// AttentionDecoder_35656818491801
// MI455X (gfx1250) — matrix operand order repaired and checked by reading, not compiled, not run on hardware
//
#include <hip/hip_runtime.h>
#include <hip/hip_bf16.h>

// AttentionDecoder for MI455X (gfx1250, wave32, WMMA).
//
// Strategy (see analysis): all GEMMs use v_wmma_f32_16x16x32_f16 with f32
// accumulate. Weights are pre-swizzled once into the WMMA B-operand lane
// layout so each WMMA needs only 2x global b128 loads per operand. The
// annotation half of the attention MLP is hoisted out of the scan. The
// serial GRU scan runs as 2 blocks x 16 batch rows (one WMMA M-tile) with
// all per-step state in 160 KB of LDS.

#define HD   256
#define BB   32
#define TT   64
#define SS   64
#define VV   32000

typedef __attribute__((ext_vector_type(16))) _Float16 v16h;
typedef __attribute__((ext_vector_type(8)))  float    v8f;
typedef __attribute__((ext_vector_type(4)))  unsigned u32x4;

union V16H {
  v16h     v;
  u32x4    q[2];
  unsigned u[8];
  _Float16 e[16];
};

// ---------------------------------------------------------------------------
// Weight swizzle into WMMA B-operand layout.
// B (32x16 f16 tile): lane = {n = lane&15, koff = (lane>>4)*8},
// element e: j=e>>1, p=e&1 -> K = kstep*32 + (j<4?0:16) + koff + 2(j&3) + p, N = ntile*16 + n.
// dst index = (((ntile*ksteps + kstep)*32 + lane)*16 + e)
// ---------------------------------------------------------------------------
__global__ void k_swizzle_single(const float* __restrict__ src,
                                 _Float16* __restrict__ dst, int K, int N) {
  int t = blockIdx.x * blockDim.x + threadIdx.x;
  if (t >= K * N) return;
  int ksteps = K >> 5;
  int h = t & 15, lane = (t >> 4) & 31, rest = t >> 9;
  int ks = rest % ksteps, ntile = rest / ksteps;
  int j = h >> 1, p = h & 1;
  int n = lane & 15, koff = (lane >> 4) * 8;
  int k = ks * 32 + ((j < 4) ? 0 : 16) + koff + 2 * (j & 3) + p;
  dst[t] = (_Float16)src[k * N + ntile * 16 + n];
}

// Same, but N is a concatenation of up to 4 column-groups of width 256,
// each group sourced from a separate (K x 256) row-major matrix.
__global__ void k_swizzle_cat(const float* __restrict__ s0, const float* __restrict__ s1,
                              const float* __restrict__ s2, const float* __restrict__ s3,
                              _Float16* __restrict__ dst, int K, int N) {
  int t = blockIdx.x * blockDim.x + threadIdx.x;
  if (t >= K * N) return;
  int ksteps = K >> 5;
  int h = t & 15, lane = (t >> 4) & 31, rest = t >> 9;
  int ks = rest % ksteps, ntile = rest / ksteps;
  int j = h >> 1, p = h & 1;
  int n = lane & 15, koff = (lane >> 4) * 8;
  int k = ks * 32 + ((j < 4) ? 0 : 16) + koff + 2 * (j & 3) + p;
  int col = ntile * 16 + n;
  int g = col >> 8, c = col & 255;
  const float* s = (g == 0) ? s0 : (g == 1) ? s1 : (g == 2) ? s2 : s3;
  dst[t] = (_Float16)s[k * 256 + c];
}

// ---------------------------------------------------------------------------
// Pack a (2048 x 256) f32 row-major matrix into the WMMA A-operand layout.
// A (16x32 f16 tile): lane = {m = lane&15, koff = (lane>>4)*8},
// element e: j=e>>1, p=e&1 -> K = kstep*32 + (j<4?0:16) + koff + 2(j&3) + p.
// ---------------------------------------------------------------------------
__global__ void k_packA(const float* __restrict__ src, _Float16* __restrict__ dst) {
  int t = blockIdx.x * blockDim.x + threadIdx.x;   // 2048*256 threads
  int h = t & 15, lane = (t >> 4) & 31;
  int ks = (t >> 9) & 7, mtile = t >> 12;
  int j = h >> 1, p = h & 1;
  int m = lane & 15, koff = (lane >> 4) * 8;
  int kk = ((j < 4) ? 0 : 16) + koff + 2 * (j & 3) + p;
  int k = ks * 32 + kk;
  dst[t] = (_Float16)src[(mtile * 16 + m) * HD + k];
}

// ---------------------------------------------------------------------------
// annP = annotations @ W1[H:2H] + b1   (2048 x 256 x 256 GEMM, f16 WMMA)
// ---------------------------------------------------------------------------
__global__ __launch_bounds__(256) void k_annproj(const _Float16* __restrict__ Apk,
                                                 const _Float16* __restrict__ Bsw,
                                                 const float* __restrict__ b1,
                                                 float* __restrict__ annP) {
  int lane = threadIdx.x & 31;
  int wave = threadIdx.x >> 5;
  int tile = blockIdx.x * 8 + wave;          // 0..2047
  int mtile = tile >> 4, ntile = tile & 15;
  v8f acc = {};
#pragma unroll
  for (int ks = 0; ks < 8; ++ks) {
    V16H a, b;
    const u32x4* pa = (const u32x4*)(Apk + (((mtile * 8 + ks) * 32 + lane) << 4));
    a.q[0] = pa[0]; a.q[1] = pa[1];
    const u32x4* pb = (const u32x4*)(Bsw + (((ntile * 8 + ks) * 32 + lane) << 4));
    b.q[0] = pb[0]; b.q[1] = pb[1];
    acc = __builtin_amdgcn_wmma_f32_16x16x32_f16(false, a.v, false, b.v,
                                                 (short)0, acc, false, false);
  }
  int n = lane & 15, hi = lane >> 4;
  int col = ntile * 16 + n;
  float bias = b1[col];
#pragma unroll
  for (int r = 0; r < 8; ++r)
    annP[(mtile * 16 + hi * 8 + r) * HD + col] = acc[r] + bias;
}

// ---------------------------------------------------------------------------
// out = hiddens @ Wout + bout   (2048 x 32000 x 256, the HBM-bound GEMM).
// Each wave owns two adjacent 16x16 tiles sharing one A operand.
// ---------------------------------------------------------------------------
__global__ __launch_bounds__(256) void k_outproj(const _Float16* __restrict__ Apk,
                                                 const _Float16* __restrict__ Bsw,
                                                 const float* __restrict__ bout,
                                                 float* __restrict__ out) {
  int lane = threadIdx.x & 31;
  int wave = threadIdx.x >> 5;
  int job = blockIdx.x * 8 + wave;           // 0..127999
  int mtile = job / 1000;
  int ntile0 = (job % 1000) * 2;
  v8f acc0 = {}, acc1 = {};
#pragma unroll
  for (int ks = 0; ks < 8; ++ks) {
    V16H a, b0, b1v;
    const u32x4* pa = (const u32x4*)(Apk + (((mtile * 8 + ks) * 32 + lane) << 4));
    a.q[0] = pa[0]; a.q[1] = pa[1];
    const u32x4* pb0 = (const u32x4*)(Bsw + (((ntile0 * 8 + ks) * 32 + lane) << 4));
    b0.q[0] = pb0[0]; b0.q[1] = pb0[1];
    const u32x4* pb1 = (const u32x4*)(Bsw + ((((ntile0 + 1) * 8 + ks) * 32 + lane) << 4));
    b1v.q[0] = pb1[0]; b1v.q[1] = pb1[1];
    acc0 = __builtin_amdgcn_wmma_f32_16x16x32_f16(false, a.v, false, b0.v,
                                                  (short)0, acc0, false, false);
    acc1 = __builtin_amdgcn_wmma_f32_16x16x32_f16(false, a.v, false, b1v.v,
                                                  (short)0, acc1, false, false);
  }
  int n = lane & 15, hi = lane >> 4;
  int col0 = ntile0 * 16 + n, col1 = col0 + 16;
  float bi0 = bout[col0], bi1 = bout[col1];
#pragma unroll
  for (int r = 0; r < 8; ++r) {
    int row = mtile * 16 + hi * 8 + r;
    out[row * VV + col0] = acc0[r] + bi0;
    out[row * VV + col1] = acc1[r] + bi1;
  }
}

// ---------------------------------------------------------------------------
// The serial scan. 2 blocks x 16 batch rows, 512 threads (16 wave32s).
// Per step: GEMM1 h@[W1q|Whz|Whr|Whh], attention (scores/softmax/context),
// GEMM2 xc@[Wiz|Wir|Wih], GRU gate math. All state in LDS (160000 B).
// ---------------------------------------------------------------------------
#define XST 776   // padded LDS stride (halves) to spread banks

__global__ __launch_bounds__(512) void k_decoder(
    const int* __restrict__ inputs, const float* __restrict__ annotations,
    const float* __restrict__ hidden_init, const float* __restrict__ emb,
    const float* __restrict__ W2, const float* __restrict__ b2,
    const float* __restrict__ biz, const float* __restrict__ bir,
    const float* __restrict__ bih, const float* __restrict__ bhz,
    const float* __restrict__ bhr, const float* __restrict__ bhh,
    const _Float16* __restrict__ Wh4, const _Float16* __restrict__ Wi3,
    const float* __restrict__ annP,
    float* __restrict__ hiddens, float* __restrict__ attnOut) {
  __shared__ float    sh_h[16 * HD];        // 16 KB   h (f32)
  __shared__ _Float16 sh_xch[16 * XST];     // 24.25KB [context | x_t | h] f16
  __shared__ float    sh_hproj[16 * 1024];  // 64 KB   [q | hz | hr | hh]
  __shared__ float    sh_iproj[16 * 768];   // 48 KB   [iz | ir | ig]
  __shared__ float    sh_sc[16 * 64];       // 4 KB    scores / attn weights

  const int tid  = threadIdx.x;
  const int lane = tid & 31;
  const int wave = tid >> 5;                // 0..15
  const int b0   = blockIdx.x * 16;
  const int bb   = tid >> 5;                // batch row owned in elementwise phases
  const int h0   = (tid & 31) * 8;          // 8 h-elements per thread

  // init hidden state
#pragma unroll
  for (int i = 0; i < 8; ++i) {
    float v = hidden_init[(b0 + bb) * HD + h0 + i];
    sh_h[bb * HD + h0 + i] = v;
    sh_xch[bb * XST + 512 + h0 + i] = (_Float16)v;
  }
  __syncthreads();

  for (int t = 0; t < TT; ++t) {
    // embedding gather -> xch[256:512]
    {
      int tok = inputs[(b0 + bb) * TT + t];
#pragma unroll
      for (int i = 0; i < 8; ++i)
        sh_xch[bb * XST + 256 + h0 + i] = (_Float16)emb[tok * HD + h0 + i];
    }
    __syncthreads();

    // GEMM1: hproj(16x1024) = hF16(16x256) @ Wh4 ; 64 N-tiles over 16 waves
    for (int job = wave; job < 64; job += 16) {
      v8f acc = {};
      int m = lane & 15, koff = (lane >> 4) * 8;
#pragma unroll
      for (int ks = 0; ks < 8; ++ks) {
        V16H a, b;
#pragma unroll
        for (int j = 0; j < 8; ++j) {
          int kk = ((j < 4) ? 0 : 16) + koff + 2 * (j & 3);
          a.u[j] = *(const unsigned*)(&sh_xch[m * XST + 512 + ks * 32 + kk]);
        }
        const u32x4* pb = (const u32x4*)(Wh4 + (((job * 8 + ks) * 32 + lane) << 4));
        b.q[0] = pb[0]; b.q[1] = pb[1];
        acc = __builtin_amdgcn_wmma_f32_16x16x32_f16(false, a.v, false, b.v,
                                                     (short)0, acc, false, false);
      }
      int n = lane & 15, hi = lane >> 4;
#pragma unroll
      for (int r = 0; r < 8; ++r)
        sh_hproj[(hi * 8 + r) * 1024 + job * 16 + n] = acc[r];
    }
    __syncthreads();

    // attention scores: relu(annP + q) . W2 + b2
    for (int id = tid; id < 16 * 64; id += 512) {
      int br = id >> 6, s = id & 63;
      const float* ap = annP + ((b0 + br) * SS + s) * HD;
      const float* qp = sh_hproj + br * 1024;     // q occupies cols [0,256)
      float acc = b2[0];
      for (int k = 0; k < HD; ++k) {
        float u = ap[k] + qp[k];
        acc += (u > 0.f ? u : 0.f) * W2[k];
      }
      sh_sc[br * 64 + s] = acc;
    }
    __syncthreads();

    // softmax over S (one thread per batch row) + emit attention output
    if (tid < 16) {
      float mx = -1e30f;
      for (int s = 0; s < 64; ++s) mx = fmaxf(mx, sh_sc[tid * 64 + s]);
      float sum = 0.f;
      for (int s = 0; s < 64; ++s) {
        float e = __expf(sh_sc[tid * 64 + s] - mx);
        sh_sc[tid * 64 + s] = e; sum += e;
      }
      float inv = 1.f / sum;
      for (int s = 0; s < 64; ++s) {
        float aw = sh_sc[tid * 64 + s] * inv;
        sh_sc[tid * 64 + s] = aw;
        attnOut[((b0 + tid) * SS + s) * TT + t] = aw;
      }
    }
    __syncthreads();

    // context = aw @ annotations -> xch[0:256]
    {
      float ctx[8] = {0.f, 0.f, 0.f, 0.f, 0.f, 0.f, 0.f, 0.f};
      const float* ann = annotations + ((b0 + bb) * SS) * HD + h0;
      for (int s = 0; s < 64; ++s) {
        float aw = sh_sc[bb * 64 + s];
#pragma unroll
        for (int i = 0; i < 8; ++i) ctx[i] += aw * ann[s * HD + i];
      }
#pragma unroll
      for (int i = 0; i < 8; ++i)
        sh_xch[bb * XST + h0 + i] = (_Float16)ctx[i];
    }
    __syncthreads();

    // GEMM2: iproj(16x768) = xcF16(16x512) @ Wi3 ; 48 N-tiles over 16 waves
    for (int job = wave; job < 48; job += 16) {
      v8f acc = {};
      int m = lane & 15, koff = (lane >> 4) * 8;
#pragma unroll
      for (int ks = 0; ks < 16; ++ks) {
        V16H a, b;
#pragma unroll
        for (int j = 0; j < 8; ++j) {
          int kk = ((j < 4) ? 0 : 16) + koff + 2 * (j & 3);
          a.u[j] = *(const unsigned*)(&sh_xch[m * XST + ks * 32 + kk]);
        }
        const u32x4* pb = (const u32x4*)(Wi3 + (((job * 16 + ks) * 32 + lane) << 4));
        b.q[0] = pb[0]; b.q[1] = pb[1];
        acc = __builtin_amdgcn_wmma_f32_16x16x32_f16(false, a.v, false, b.v,
                                                     (short)0, acc, false, false);
      }
      int n = lane & 15, hi = lane >> 4;
#pragma unroll
      for (int r = 0; r < 8; ++r)
        sh_iproj[(hi * 8 + r) * 768 + job * 16 + n] = acc[r];
    }
    __syncthreads();

    // GRU gate math + h update
    {
#pragma unroll
      for (int i = 0; i < 8; ++i) {
        int h = h0 + i;
        float hz = sh_hproj[bb * 1024 + 256 + h];
        float hr = sh_hproj[bb * 1024 + 512 + h];
        float hh = sh_hproj[bb * 1024 + 768 + h];
        float iz = sh_iproj[bb * 768 + h];
        float ir = sh_iproj[bb * 768 + 256 + h];
        float ig = sh_iproj[bb * 768 + 512 + h];
        float z = 1.f / (1.f + __expf(-(iz + hz + biz[h] + bhz[h])));
        float r = 1.f / (1.f + __expf(-(ir + hr + bir[h] + bhr[h])));
        float g = tanhf(ig + bih[h] + r * (hh + bhh[h]));
        float hn = (1.f - z) * g + z * sh_h[bb * HD + h];
        sh_h[bb * HD + h] = hn;
        sh_xch[bb * XST + 512 + h] = (_Float16)hn;
        hiddens[((b0 + bb) * TT + t) * HD + h] = hn;
      }
    }
    __syncthreads();
  }
}

// ---------------------------------------------------------------------------
extern "C" void kernel_launch(void* const* d_in, const int* in_sizes, int n_in,
                              void* d_out, int out_size, void* d_ws, size_t ws_size,
                              hipStream_t stream) {
  (void)in_sizes; (void)n_in; (void)out_size; (void)ws_size;
  const int*   inputs      = (const int*)  d_in[0];
  const float* annotations = (const float*)d_in[1];
  const float* hidden_init = (const float*)d_in[2];
  const float* emb         = (const float*)d_in[3];
  const float* W1          = (const float*)d_in[4];
  const float* b1          = (const float*)d_in[5];
  const float* W2          = (const float*)d_in[6];
  const float* b2          = (const float*)d_in[7];
  const float* Wiz         = (const float*)d_in[8];
  const float* biz         = (const float*)d_in[9];
  const float* Wir         = (const float*)d_in[10];
  const float* bir         = (const float*)d_in[11];
  const float* Wih         = (const float*)d_in[12];
  const float* bih         = (const float*)d_in[13];
  const float* Whz         = (const float*)d_in[14];
  const float* bhz         = (const float*)d_in[15];
  const float* Whr         = (const float*)d_in[16];
  const float* bhr         = (const float*)d_in[17];
  const float* Whh         = (const float*)d_in[18];
  const float* bhh         = (const float*)d_in[19];
  const float* Wout        = (const float*)d_in[20];
  const float* bout        = (const float*)d_in[21];

  // workspace layout (~24.1 MB total)
  char* ws = (char*)d_ws;
  _Float16* WoutSwz = (_Float16*)(ws + 0);          // 16,384,000 B
  _Float16* Wh4Swz  = (_Float16*)(ws + 16384000);   //    524,288 B
  _Float16* Wi3Swz  = (_Float16*)(ws + 16908288);   //    786,432 B
  _Float16* W1aSwz  = (_Float16*)(ws + 17694720);   //    131,072 B
  _Float16* annPk   = (_Float16*)(ws + 17825792);   //  1,048,576 B
  float*    annP    = (float*)   (ws + 18874368);   //  2,097,152 B
  float*    hiddens = (float*)   (ws + 20971520);   //  2,097,152 B
  _Float16* hidPk   = (_Float16*)(ws + 23068672);   //  1,048,576 B

  float* out     = (float*)d_out;                   // (B,T,V)
  float* attnOut = out + (size_t)BB * TT * VV;      // (B,S,T)

  // one-time weight prep (runs every call for determinism; cheap vs GEMM)
  k_swizzle_single<<<32000, 256, 0, stream>>>(Wout, WoutSwz, 256, VV);
  k_swizzle_single<<<256,   256, 0, stream>>>(W1 + 256 * 256, W1aSwz, 256, 256);
  k_swizzle_cat<<<1024, 256, 0, stream>>>(W1, Whz, Whr, Whh, Wh4Swz, 256, 1024);
  k_swizzle_cat<<<1536, 256, 0, stream>>>(Wiz, Wir, Wih, Wih, Wi3Swz, 512, 768);

  // hoisted annotation projection
  k_packA<<<2048, 256, 0, stream>>>(annotations, annPk);
  k_annproj<<<256, 256, 0, stream>>>(annPk, W1aSwz, b1, annP);

  // serial attention-GRU scan
  k_decoder<<<2, 512, 0, stream>>>(inputs, annotations, hidden_init, emb, W2, b2,
                                   biz, bir, bih, bhz, bhr, bhh,
                                   Wh4Swz, Wi3Swz, annP, hiddens, attnOut);

  // HBM-bound output projection
  k_packA<<<2048, 256, 0, stream>>>(hiddens, hidPk);
  k_outproj<<<16000, 256, 0, stream>>>(hidPk, WoutSwz, bout, out);
}
